// GraphEncoder_35605278884121
// MI455X (gfx1250) — matrix operand order repaired and checked by reading, not compiled, not run on hardware
//
#include <hip/hip_runtime.h>
#include <hip/hip_bf16.h>

typedef __attribute__((ext_vector_type(16))) __bf16 v16bf;
typedef __attribute__((ext_vector_type(8)))  __bf16 bf16x8;
typedef __attribute__((ext_vector_type(8)))  float  v8f;

#define D 128
#define ROWS_BLK 64   // rows of the node matrix per block (4 M-tiles of 16)

__device__ __forceinline__ __bf16 f2bf(float f) {
    unsigned u = __builtin_bit_cast(unsigned, f);
    unsigned r = (u + 0x7FFFu + ((u >> 16) & 1u)) >> 16;   // round-to-nearest-even
    unsigned short s = (unsigned short)r;
    return __builtin_bit_cast(__bf16, s);
}

// A-fragment: 16x32 bf16 tile (M x K), row-major in LDS with stride 128.
// ISA 7.12.2 layout: lanes 0-15 -> M=lane, K in {kk+0..7, kk+16..23};
//                    lanes 16-31 -> M=lane-16, K in {kk+8..15, kk+24..31}.
__device__ __forceinline__ v16bf load_afrag(const __bf16* __restrict__ tile,
                                            int lane, int kk) {
    int lo  = (lane & 16) ? 8 : 0;
    int row = lane & 15;
    const bf16x8* p0 = (const bf16x8*)(tile + row * D + kk + lo);
    const bf16x8* p1 = (const bf16x8*)(tile + row * D + kk + 16 + lo);
    union { v16bf v; bf16x8 h[2]; } u;
    u.h[0] = *p0;
    u.h[1] = *p1;
    return u.v;
}

// B-fragment: 32x16 bf16 tile (K x N). Weights stored TRANSPOSED in LDS as
// W_T[n][k] (n-major, stride 128), so each lane reads two runs of 8 K values.
// lanes 0-15 -> N=lane, K in {kk+0..7, kk+16..23}; lanes 16-31 -> N=lane-16, K in {kk+8..15, kk+24..31}.
__device__ __forceinline__ v16bf load_bfrag(const __bf16* __restrict__ wT,
                                            int lane, int kk, int ntile) {
    int n  = (ntile << 4) + (lane & 15);
    int kb = kk + ((lane & 16) ? 8 : 0);
    const bf16x8* p = (const bf16x8*)(wT + n * D + kb);
    union { v16bf v; bf16x8 h[2]; } u;
    u.h[0] = p[0];
    u.h[1] = p[2];
    return u.v;
}

// ---------------------------------------------------------------- kernels ---

// h[i][:] = emb[x[i]][:]
__global__ __launch_bounds__(256) void gather_kernel(const int* __restrict__ x,
                                                     const float* __restrict__ emb,
                                                     float* __restrict__ h, int n) {
    int i = blockIdx.x * blockDim.x + threadIdx.x;   // over n*32 float4 chunks
    if (i >= n * 32) return;
    int node = i >> 5, c4 = (i & 31) << 2;
    int src = x[node];
    *(float4*)(h + (size_t)node * D + c4) =
        *(const float4*)(emb + (size_t)src * D + c4);
}

// Convert fp32 weights to bf16 and transpose: WT[l][n][k] = W[l][k][n]
__global__ __launch_bounds__(256) void wconv_kernel(const float* __restrict__ Wa,
                                                    const float* __restrict__ Wb,
                                                    __bf16* __restrict__ WaT,
                                                    __bf16* __restrict__ WbT,
                                                    int total) {
    int i = blockIdx.x * blockDim.x + threadIdx.x;
    if (i >= total) return;                 // total = L*128*128
    int l = i / (D * D);
    int n = (i / D) % D;
    int k = i % D;
    size_t s = (size_t)l * D * D + (size_t)k * D + n;
    WaT[i] = f2bf(Wa[s]);
    WbT[i] = f2bf(Wb[s]);
}

// z = h (float4 copy)
__global__ __launch_bounds__(256) void copy_kernel(const float* __restrict__ h,
                                                   float* __restrict__ z, int n4) {
    int i = blockIdx.x * blockDim.x + threadIdx.x;
    if (i < n4) ((float4*)z)[i] = ((const float4*)h)[i];
}

// z[dst] += h[src] for each edge, 32 threads per edge (float4 granularity)
__global__ __launch_bounds__(256) void scatter_kernel(const int* __restrict__ ei,
                                                      const float* __restrict__ h,
                                                      float* __restrict__ z,
                                                      int nedges) {
    long long i = (long long)blockIdx.x * blockDim.x + threadIdx.x;
    if (i >= (long long)nedges * 32) return;
    int e  = (int)(i >> 5);
    int c4 = ((int)i & 31) << 2;
    int s = ei[e];
    int d = ei[nedges + e];
    const float4 v = *(const float4*)(h + (size_t)s * D + c4);
    float* p = z + (size_t)d * D + c4;
    atomicAdd(p + 0, v.x);
    atomicAdd(p + 1, v.y);
    atomicAdd(p + 2, v.z);
    atomicAdd(p + 3, v.w);
}

// Fused 2-layer MLP: h_out = relu(relu(z@Wa + ba) @ Wb + bb)
// 8 waves/block; wave w owns N-tile w; 4 M-tiles per wave (64 rows/block).
__global__ __launch_bounds__(256) void mlp_kernel(const float* __restrict__ z,
                                                  const __bf16* __restrict__ WaT,
                                                  const float* __restrict__ ba,
                                                  const __bf16* __restrict__ WbT,
                                                  const float* __restrict__ bb,
                                                  float* __restrict__ hout,
                                                  int layer, int nnodes) {
    extern __shared__ __align__(16) char smem[];
    __bf16* sZ = (__bf16*)smem;                 // 64*128 bf16 = 16 KB
    __bf16* sT = sZ + ROWS_BLK * D;             // 64*128 bf16 = 16 KB
    __bf16* sW = sT + ROWS_BLK * D;             // 128*128 bf16 = 32 KB (reused)

    const int tid  = threadIdx.x;
    const int lane = tid & 31;
    const int nt   = tid >> 5;                  // wave id == N tile
    const int row0 = blockIdx.x * ROWS_BLK;

    // Stage z -> bf16 LDS (zero-pad past nnodes)
    for (int i = tid; i < ROWS_BLK * D / 4; i += 256) {
        int r  = i >> 5;
        int c4 = (i & 31) << 2;
        int row = row0 + r;
        float4 v = make_float4(0.f, 0.f, 0.f, 0.f);
        if (row < nnodes) v = *(const float4*)(z + (size_t)row * D + c4);
        __bf16* d = sZ + r * D + c4;
        d[0] = f2bf(v.x); d[1] = f2bf(v.y); d[2] = f2bf(v.z); d[3] = f2bf(v.w);
    }
    // Stage Wa^T (bf16, already transposed in global)
    {
        const float4* wsrc = (const float4*)(WaT + (size_t)layer * D * D);
        float4* wdst = (float4*)sW;
        for (int i = tid; i < D * D / 8; i += 256) wdst[i] = wsrc[i];
    }
    __syncthreads();

    // ---- GEMM 1: T = relu(Z @ Wa + ba) ----
    const float biasa = ba[layer * D + nt * 16 + (lane & 15)];
    {
        v16bf bfr[4];
        for (int k4 = 0; k4 < 4; ++k4) bfr[k4] = load_bfrag(sW, lane, k4 * 32, nt);
        for (int mt = 0; mt < 4; ++mt) {
            v8f acc = {0.f, 0.f, 0.f, 0.f, 0.f, 0.f, 0.f, 0.f};
            const __bf16* tile = sZ + mt * 16 * D;
            for (int k4 = 0; k4 < 4; ++k4) {
                v16bf a = load_afrag(tile, lane, k4 * 32);
                acc = __builtin_amdgcn_wmma_f32_16x16x32_bf16(
                    false, a, false, bfr[k4], (short)0, acc, false, false);
            }
            int mbase = mt * 16 + ((lane & 16) ? 8 : 0);
            int col   = nt * 16 + (lane & 15);
            for (int r = 0; r < 8; ++r)
                sT[(mbase + r) * D + col] = f2bf(fmaxf(acc[r] + biasa, 0.f));
        }
    }
    __syncthreads();

    // Restage Wb^T over the same LDS region
    {
        const float4* wsrc = (const float4*)(WbT + (size_t)layer * D * D);
        float4* wdst = (float4*)sW;
        for (int i = tid; i < D * D / 8; i += 256) wdst[i] = wsrc[i];
    }
    __syncthreads();

    // ---- GEMM 2: H = relu(T @ Wb + bb) ----
    const float biasb = bb[layer * D + nt * 16 + (lane & 15)];
    {
        v16bf bfr[4];
        for (int k4 = 0; k4 < 4; ++k4) bfr[k4] = load_bfrag(sW, lane, k4 * 32, nt);
        for (int mt = 0; mt < 4; ++mt) {
            v8f acc = {0.f, 0.f, 0.f, 0.f, 0.f, 0.f, 0.f, 0.f};
            const __bf16* tile = sT + mt * 16 * D;
            for (int k4 = 0; k4 < 4; ++k4) {
                v16bf a = load_afrag(tile, lane, k4 * 32);
                acc = __builtin_amdgcn_wmma_f32_16x16x32_bf16(
                    false, a, false, bfr[k4], (short)0, acc, false, false);
            }
            int mbase = mt * 16 + ((lane & 16) ? 8 : 0);
            int col   = nt * 16 + (lane & 15);
            for (int r = 0; r < 8; ++r) {
                int row = row0 + mbase + r;
                if (row < nnodes)
                    hout[(size_t)row * D + col] = fmaxf(acc[r] + biasb, 0.f);
            }
        }
    }
}

__global__ void zero_kernel(float* __restrict__ p, int n) {
    int i = blockIdx.x * blockDim.x + threadIdx.x;
    if (i < n) p[i] = 0.f;
}

// pooled[c] += sum over a 256-row chunk of h[:,c]
__global__ __launch_bounds__(128) void reduce_kernel(const float* __restrict__ h,
                                                     float* __restrict__ pooled,
                                                     int nnodes) {
    int c  = threadIdx.x;
    int r0 = blockIdx.x * 256;
    int r1 = min(r0 + 256, nnodes);
    float s = 0.f;
    for (int r = r0; r < r1; ++r) s += h[(size_t)r * D + c];
    atomicAdd(&pooled[c], s);
}

// out = pooled @ Wlin + blin   (1x128 @ 128x128)
__global__ __launch_bounds__(128) void final_kernel(const float* __restrict__ pooled,
                                                    const float* __restrict__ Wlin,
                                                    const float* __restrict__ blin,
                                                    float* __restrict__ out) {
    int o = threadIdx.x;
    float s = blin[o];
    for (int d = 0; d < D; ++d) s += pooled[d] * Wlin[d * D + o];
    out[o] = s;
}

// --------------------------------------------------------------- launcher ---

extern "C" void kernel_launch(void* const* d_in, const int* in_sizes, int n_in,
                              void* d_out, int out_size, void* d_ws, size_t ws_size,
                              hipStream_t stream) {
    const int*   x    = (const int*)d_in[0];
    const int*   ei   = (const int*)d_in[1];
    const float* emb  = (const float*)d_in[2];
    const float* Wa   = (const float*)d_in[3];
    const float* ba   = (const float*)d_in[4];
    const float* Wb   = (const float*)d_in[5];
    const float* bb   = (const float*)d_in[6];
    const float* Wlin = (const float*)d_in[7];
    const float* blin = (const float*)d_in[8];
    float* out = (float*)d_out;

    const int nnodes = in_sizes[0];
    const int nedges = in_sizes[1] / 2;
    const int L      = in_sizes[3] / (D * D);

    // Workspace layout
    char* ws = (char*)d_ws;
    size_t featBytes = (size_t)nnodes * D * sizeof(float);
    float*  h_buf  = (float*)ws;                               ws += featBytes;
    float*  z_buf  = (float*)ws;                               ws += featBytes;
    __bf16* WaT    = (__bf16*)ws;  ws += (size_t)L * D * D * sizeof(__bf16);
    __bf16* WbT    = (__bf16*)ws;  ws += (size_t)L * D * D * sizeof(__bf16);
    float*  pooled = (float*)ws;

    const int smem = (ROWS_BLK * D + ROWS_BLK * D + D * D) * (int)sizeof(__bf16); // 64 KB

    // h = emb[x]
    {
        int tot = nnodes * 32;
        gather_kernel<<<(tot + 255) / 256, 256, 0, stream>>>(x, emb, h_buf, nnodes);
    }
    // weight convert+transpose (once)
    {
        int tot = L * D * D;
        wconv_kernel<<<(tot + 255) / 256, 256, 0, stream>>>(Wa, Wb, WaT, WbT, tot);
    }

    int mlp_blocks = (nnodes + ROWS_BLK - 1) / ROWS_BLK;
    for (int l = 0; l < L; ++l) {
        int n4 = nnodes * (D / 4);
        copy_kernel<<<(n4 + 255) / 256, 256, 0, stream>>>(h_buf, z_buf, n4);
        long long st = (long long)nedges * 32;
        scatter_kernel<<<(int)((st + 255) / 256), 256, 0, stream>>>(ei, h_buf, z_buf, nedges);
        mlp_kernel<<<mlp_blocks, 256, smem, stream>>>(z_buf, WaT, ba, WbT, bb,
                                                      h_buf, l, nnodes);
    }

    zero_kernel<<<1, 128, 0, stream>>>(pooled, D);
    reduce_kernel<<<(nnodes + 255) / 256, 128, 0, stream>>>(h_buf, pooled, nnodes);
    final_kernel<<<1, 128, 0, stream>>>(pooled, Wlin, blin, out);
}
